// RGCN_65317862637841
// MI455X (gfx1250) — matrix operand order repaired and checked by reading, not compiled, not run on hardware
//
#include <hip/hip_runtime.h>
#include <hip/hip_bf16.h>

// ---------- problem constants (fixed by the reference) ----------
#define N_NODES  20000
#define N_EDGES  300000
#define DIM      256
#define NUM_RELS 8
#define NUM_BASES 4

typedef __bf16 bf16;
typedef __attribute__((ext_vector_type(16))) __bf16 v16bf;
typedef __attribute__((ext_vector_type(4)))  __bf16 v4bf;
typedef __attribute__((ext_vector_type(8)))  float  v8f;

union FragBF16 { uint4 u[2]; v16bf v; };

// ---------------------------------------------------------------
// Kernel 1: Wt[r][o][i] = sum_b comp[r][b] * basis[b][i][o]  (bf16, N-major)
// 8*256*256 = 524288 outputs, 4 fma each.
__global__ __launch_bounds__(256)
void rgcn_build_wt(const float* __restrict__ comp, const float* __restrict__ basis,
                   bf16* __restrict__ wt) {
    int idx = blockIdx.x * 256 + threadIdx.x;     // < 524288
    int i = idx & 255;
    int o = (idx >> 8) & 255;
    int r = idx >> 16;
    float acc = 0.f;
#pragma unroll
    for (int b = 0; b < NUM_BASES; ++b)
        acc = fmaf(comp[r * NUM_BASES + b], basis[(size_t)b * 65536 + i * 256 + o], acc);
    wt[(size_t)r * 65536 + o * 256 + i] = (bf16)acc;   // transposed store
}

// Kernel 2: Lt[o][i] = loop_weight[i][o]  (bf16, N-major)
__global__ __launch_bounds__(256)
void rgcn_build_lt(const float* __restrict__ lw, bf16* __restrict__ lt) {
    int idx = blockIdx.x * 256 + threadIdx.x;     // < 65536
    int i = idx & 255;
    int o = idx >> 8;
    lt[o * 256 + i] = (bf16)lw[i * 256 + o];
}

// Kernel 3: x (f32) -> xb (bf16), 4 elements/thread
__global__ __launch_bounds__(256)
void rgcn_conv_x(const float* __restrict__ x, bf16* __restrict__ xb) {
    int idx = blockIdx.x * 256 + threadIdx.x;     // < 1,280,000
    const float4 f = ((const float4*)x)[idx];
    v4bf o;
    o[0] = (bf16)f.x; o[1] = (bf16)f.y; o[2] = (bf16)f.z; o[3] = (bf16)f.w;
    *(v4bf*)(xb + (size_t)idx * 4) = o;
}

// ---------------------------------------------------------------
// Wave-level 16x64 bf16 GEMM tile: C[16,64] += A[16,256] * B[256,64]
// A row-major [M][K] bf16, Bt is B transposed: [N][K] bf16 row-major.
// Matches CDNA5 WMMA 16-bit A layout (lane half-wave K offset 8, chunks at
// K+0 and K+16) and B layout (lane = column, same K order as A).
__device__ __forceinline__
void wave_gemm_16x64(const bf16* __restrict__ A_m0,   // A + m0*256
                     const bf16* __restrict__ Bt,     // [256][256] transposed
                     int n0w, int lane, v8f acc[4]) {
    const int lh  = lane >> 4;        // 0 or 1 (half-wave)
    const int l15 = lane & 15;
    const bf16* Arow = A_m0 + (size_t)l15 * 256 + lh * 8;
#pragma unroll
    for (int kk = 0; kk < 256; kk += 32) {
        FragBF16 a;
        a.u[0] = *(const uint4*)(Arow + kk);        // K = kk+lh*8 .. +7
        a.u[1] = *(const uint4*)(Arow + kk + 16);   // K = kk+16+lh*8 .. +7
#pragma unroll
        for (int t = 0; t < 4; ++t) {
            const int n = n0w + t * 16 + l15;
            const bf16* bp = Bt + (size_t)n * 256 + kk + lh * 8;
            FragBF16 b;
            b.u[0] = *(const uint4*)(bp);
            b.u[1] = *(const uint4*)(bp + 16);
            acc[t] = __builtin_amdgcn_wmma_f32_16x16x32_bf16(
                false, a.v, false, b.v, (short)0, acc[t], false, false);
        }
    }
}

// Kernel 4: out[m][n] = x·loop_weight + bias   (f32, initializes d_out)
// grid (1250,1), block 128 (4 waves), wave w handles N block w*64.
__global__ __launch_bounds__(128)
void rgcn_gemm_loop(const bf16* __restrict__ xb, const bf16* __restrict__ lt,
                    const float* __restrict__ bias, float* __restrict__ out) {
    const int lane = threadIdx.x & 31;
    const int wave = threadIdx.x >> 5;
    const int m0   = blockIdx.x * 16;
    const int n0w  = wave * 64;
    v8f acc[4] = {};
    wave_gemm_16x64(xb + (size_t)m0 * 256, lt, n0w, lane, acc);
    const int rowb = m0 + (lane >> 4) * 8;
    const int l15  = lane & 15;
#pragma unroll
    for (int t = 0; t < 4; ++t) {
        const int col = n0w + t * 16 + l15;
        const float bv = bias[col];
#pragma unroll
        for (int v = 0; v < 8; ++v)
            out[(size_t)(rowb + v) * 256 + col] = acc[t][v] + bv;
    }
}

// Kernel 5: H[r][m][n] = x·W_r  (bf16). grid (1250, 8), block 128.
__global__ __launch_bounds__(128)
void rgcn_gemm_h(const bf16* __restrict__ xb, const bf16* __restrict__ wt,
                 bf16* __restrict__ H) {
    const int lane = threadIdx.x & 31;
    const int wave = threadIdx.x >> 5;
    const int m0   = blockIdx.x * 16;
    const int r    = blockIdx.y;
    const int n0w  = wave * 64;
    v8f acc[4] = {};
    wave_gemm_16x64(xb + (size_t)m0 * 256, wt + (size_t)r * 65536, n0w, lane, acc);
    bf16* Hr = H + (size_t)r * N_NODES * 256;
    const int rowb = m0 + (lane >> 4) * 8;
    const int l15  = lane & 15;
#pragma unroll
    for (int t = 0; t < 4; ++t) {
        const int col = n0w + t * 16 + l15;
#pragma unroll
        for (int v = 0; v < 8; ++v)
            Hr[(size_t)(rowb + v) * 256 + col] = (bf16)acc[t][v];
    }
}

// Kernel 6: per-edge gather from H + atomic scatter-add into out.
// One wave per edge; lane l handles columns j*32 + l (coalesced loads & atomics).
// unsafeAtomicAdd guarantees the hardware global_atomic_add_f32 path
// (no CAS retry loop) — the scatter phase is the dominant cost.
__global__ __launch_bounds__(256)
void rgcn_edge_scatter(const int* __restrict__ src, const int* __restrict__ dst,
                       const int* __restrict__ et, const bf16* __restrict__ H,
                       float* __restrict__ out, int nEdges) {
    const int e = blockIdx.x * 8 + (threadIdx.x >> 5);
    if (e >= nEdges) return;
    const int lane = threadIdx.x & 31;
    const int s = src[e];
    const int d = dst[e];
    const int r = et[e];
    const bf16* h = H + ((size_t)r * N_NODES + (size_t)s) * 256;
    float* o = out + (size_t)d * 256;
#pragma unroll
    for (int j = 0; j < 8; ++j) {
        const int c = j * 32 + lane;
        unsafeAtomicAdd(o + c, (float)h[c]);
    }
}

// ---------------------------------------------------------------
extern "C" void kernel_launch(void* const* d_in, const int* in_sizes, int n_in,
                              void* d_out, int out_size, void* d_ws, size_t ws_size,
                              hipStream_t stream) {
    const float* x     = (const float*)d_in[0];
    const int*   src   = (const int*)  d_in[1];
    const int*   dst   = (const int*)  d_in[2];
    const int*   et    = (const int*)  d_in[3];
    const float* basis = (const float*)d_in[4];
    const float* comp  = (const float*)d_in[5];
    const float* lw    = (const float*)d_in[6];
    const float* bias  = (const float*)d_in[7];
    float* out = (float*)d_out;
    const int nE = in_sizes[1];

    // workspace layout (bytes), all 256-aligned
    char* ws = (char*)d_ws;
    bf16* wt = (bf16*)(ws);                       //  1,048,576 B : 8x256x256 bf16
    bf16* lt = (bf16*)(ws + 1048576);             //    131,072 B : 256x256 bf16
    bf16* xb = (bf16*)(ws + 1179648);             // 10,240,000 B : 20000x256 bf16
    bf16* H  = (bf16*)(ws + 11419648);            // 81,920,000 B : 8x20000x256 bf16

    rgcn_build_wt<<<2048, 256, 0, stream>>>(comp, basis, wt);
    rgcn_build_lt<<<256, 256, 0, stream>>>(lw, lt);
    rgcn_conv_x<<<(N_NODES * DIM / 4 + 255) / 256, 256, 0, stream>>>(x, xb);
    rgcn_gemm_loop<<<dim3(N_NODES / 16, 1), 128, 0, stream>>>(xb, lt, bias, out);
    rgcn_gemm_h<<<dim3(N_NODES / 16, NUM_RELS), 128, 0, stream>>>(xb, wt, H);
    rgcn_edge_scatter<<<(nE + 7) / 8, 256, 0, stream>>>(src, dst, et, H, out, nE);
}
